// ConnectedLossV4_83760452206649
// MI455X (gfx1250) — compile-verified, matrix operand order checked by reading, not run on hardware
//
#include <hip/hip_runtime.h>
#include <stdint.h>

// ---------------- problem constants ----------------
#define HH 512
#define WW 512
#define BATCH 4
#define NC 8                    // pred channels / class alphabet
#define HW (HH*WW)              // 262144
#define TOT (BATCH*HW)          // 1048576
#define WP 514                  // padded width/height (1-px zero halo)
#define PLANEP (WP*WP)          // 264196
#define CC_PASSES 12
#define CC_INNER 48
#define MED_BLOCKS 512          // 512*256 threads = 4096 waves; 8 chunks of 32 px each

// ---------------- accumulator layout (int32 slots inside ws) ----------------
#define A_NCOMP 0               // [8] per-class component count
#define A_NPIX  8               // [8] per-class pixel count
#define A_CNT_T 16              // [8] per-target-label pixel count
#define A_B1    24              // [8] radix-select bucket 1
#define A_RK1   32              // [8] residual rank
#define A_B2    40              // [8]
#define A_RK2   48              // [8]
#define A_MED   56              // [8] median float bits
#define A_CV    64              // [8] per-class constant c_v (float)
#define A_BG    73              // [4] bg loss accumulators (float)
#define A_TSUM  80              // [28] per-target accumulators (float): [-,Sftv,Sf,Stv]
#define A_HA    128             // 7*2048 histogram, bits[31:21]
#define A_HB    (A_HA + 7*2048) // 7*2048 histogram, bits[20:10]
#define A_HC    (A_HB + 7*2048) // 7*1024 histogram, bits[9:0]
#define A_END   (A_HC + 7*1024)

typedef unsigned int v4u __attribute__((ext_vector_type(4)));
typedef int          v8i __attribute__((ext_vector_type(8)));
typedef int          v4i __attribute__((ext_vector_type(4)));
typedef _Float16     v16h __attribute__((ext_vector_type(16)));
typedef float        v8f __attribute__((ext_vector_type(8)));

__device__ __forceinline__ float wave_red(float v) {
  #pragma unroll
  for (int o = 16; o > 0; o >>= 1) v += __shfl_down(v, o, 32);
  return v;
}

// ---------------- zero the accumulator region ----------------
__global__ void cl_zero(unsigned* acc) {
  int i = blockIdx.x * 256 + threadIdx.x;
  if (i < A_END) acc[i] = 0u;
}

// ---------------- argmax + packed CC seed on padded plane ----------------
// state = (class<<24) | (flat_index+1) for class!=0 else 0; halo = 0
__global__ void cl_init(const float* __restrict__ pred, unsigned* __restrict__ st) {
  int i = blockIdx.x * 256 + threadIdx.x;
  if (i >= BATCH * PLANEP) return;
  int b = i / PLANEP; int r = i - b * PLANEP;
  int y = r / WP;     int x = r - y * WP;
  unsigned s = 0u;
  if (y > 0 && y < WP - 1 && x > 0 && x < WP - 1) {
    int uy = y - 1, ux = x - 1;
    const float* p = pred + (size_t)b * NC * HW + (size_t)uy * WW + ux;
    float best = p[0]; int bc = 0;
    #pragma unroll
    for (int c = 1; c < NC; ++c) {
      float v = p[(size_t)c * HW];
      if (v > best) { best = v; bc = c; }   // strict > : first max wins (argmax)
    }
    if (bc) {
      int flat = b * HW + uy * WW + ux;
      s = ((unsigned)bc << 24) | (unsigned)(flat + 1);
    }
  }
  st[i] = s;
}

// ---------------- one CC pass: TDM tile load -> LDS min-propagation ----------------
__global__ void cl_cc(unsigned* __restrict__ st) {
  __shared__ unsigned tile[34 * 34];
  __shared__ int conv;
  const int tx = blockIdx.x, ty = blockIdx.y, b = blockIdx.z;
  const int tid = threadIdx.x;

  unsigned* gsrc = st + ((size_t)b * WP + (size_t)ty * 32) * WP + (size_t)tx * 32;

#if __has_builtin(__builtin_amdgcn_tensor_load_to_lds)
  if (tid < 32) {  // wave 0 issues the DMA (TDM ignores EXEC; one issue per wave)
    uint64_t ga = (uint64_t)(uintptr_t)gsrc;
    unsigned lds = (unsigned)(uintptr_t)(&tile[0]);   // low 32 bits = LDS offset
    v4u g0; v8i g1; v4i g2, g3; v8i g4;
    g0[0] = 1u;                                        // count = 1 valid descriptor
    g0[1] = lds;                                       // lds_addr
    g0[2] = (unsigned)(ga & 0xffffffffu);              // global_addr lo
    g0[3] = (unsigned)((ga >> 32) & 0x1ffffffu) | (2u << 30);  // addr hi | type=2
    g1[0] = (int)(2u << 16);                           // data_size = 4 bytes
    g1[1] = (int)((unsigned)WP << 16);                 // tensor_dim0[15:0]
    g1[2] = (int)((unsigned)WP << 16);                 // dim0 hi=0 | tensor_dim1[15:0]
    g1[3] = (int)(34u << 16);                          // dim1 hi=0 | tile_dim0 = 34
    g1[4] = (int)34u;                                  // tile_dim1 = 34, tile_dim2 = 0
    g1[5] = (int)WP;                                   // tensor_dim0_stride lo = 514
    g1[6] = 0; g1[7] = 0;                              // stride hi / dim1_stride = 0
    g2 = (v4i){0, 0, 0, 0};
    g3 = (v4i){0, 0, 0, 0};
    g4 = (v8i){0, 0, 0, 0, 0, 0, 0, 0};
    __builtin_amdgcn_tensor_load_to_lds(g0, g1, g2, g3, g4, 0);
    __builtin_amdgcn_s_wait_tensorcnt(0);
  }
#else
  for (int p = tid; p < 34 * 34; p += 256) {
    int y = p / 34, x = p - y * 34;
    tile[p] = gsrc[(size_t)y * WP + x];
  }
#endif
  __syncthreads();
  if (tid == 0) conv = 0;
  __syncthreads();

  // min-propagation to local convergence (halo is read-only within the pass)
  for (int it = 0; it < CC_INNER; ++it) {
    unsigned nv[4]; int ch = 0;
    #pragma unroll
    for (int k = 0; k < 4; ++k) {
      int p = tid + k * 256; int y = p >> 5, x = p & 31;
      int li = (y + 1) * 34 + (x + 1);
      unsigned s = tile[li], m = s;
      if (s >> 24) {
        unsigned n;
        n = tile[li - 34]; if (!((n ^ s) >> 24)) m = min(m, n);
        n = tile[li + 34]; if (!((n ^ s) >> 24)) m = min(m, n);
        n = tile[li - 1];  if (!((n ^ s) >> 24)) m = min(m, n);
        n = tile[li + 1];  if (!((n ^ s) >> 24)) m = min(m, n);
      }
      nv[k] = m; ch |= (m != s);
    }
    __syncthreads();
    #pragma unroll
    for (int k = 0; k < 4; ++k) {
      int p = tid + k * 256; int y = p >> 5, x = p & 31;
      tile[(y + 1) * 34 + (x + 1)] = nv[k];
    }
    if (ch) atomicOr(&conv, 1);
    __syncthreads();
    int done = (conv == 0);
    __syncthreads();            // everyone read conv before reset
    if (tid == 0) conv = 0;
    if (done) break;            // uniform
  }
  __syncthreads();
  // write interior back (each pixel owned by exactly one block)
  #pragma unroll
  for (int k = 0; k < 4; ++k) {
    int p = tid + k * 256; int y = p >> 5, x = p & 31;
    st[((size_t)b * WP + ty * 32 + y + 1) * WP + tx * 32 + x + 1] =
        tile[(y + 1) * 34 + (x + 1)];
  }
}

// ---------------- per-class pixel / component-representative counts ----------------
__global__ void cl_stats(const unsigned* __restrict__ st, unsigned* acc) {
  int i = blockIdx.x * 256 + threadIdx.x;
  int b = i >> 18; int r = i & (HW - 1); int y = r >> 9; int x = r & 511;
  unsigned s = st[((size_t)b * WP + y + 1) * WP + x + 1];
  unsigned cls = s >> 24;
  if (cls) {
    atomicAdd(&acc[A_NPIX + cls], 1u);
    if ((s & 0xffffffu) == (unsigned)(i + 1))      // component representative
      atomicAdd(&acc[A_NCOMP + cls], 1u);
  }
}

// ---------------- last_i recurrence -> per-class constant c_v ----------------
__global__ void cl_scalars(unsigned* acc) {
  if (threadIdx.x || blockIdx.x) return;
  float* accf = (float*)acc;
  unsigned last_i = 1u;
  for (int v = 1; v < NC; ++v) {
    unsigned np = acc[A_NPIX + v], nc = acc[A_NCOMP + v];
    bool present = np > 0u;
    accf[A_CV + v] = present ? ((float)nc * (float)last_i) : 0.0f;
    if (present) last_i += nc + ((np < (unsigned)TOT) ? 1u : 0u);  // + has_bg
  }
}

// ---------------- placeholder field (exact float accumulation order) ----------------
__global__ void cl_place(const unsigned* __restrict__ st,
                         const unsigned* __restrict__ acc,
                         float* __restrict__ ph) {
  int i = blockIdx.x * 256 + threadIdx.x;
  const float* accf = (const float*)acc;
  int b = i >> 18; int r = i & (HW - 1); int y = r >> 9; int x = r & 511;
  unsigned s = st[((size_t)b * WP + y + 1) * WP + x + 1];
  unsigned cls = s >> 24;
  float lab = (float)(s & 0xffffffu);
  float v = 0.f;
  #pragma unroll
  for (int c = 1; c < NC; ++c)
    v += ((cls == (unsigned)c) ? lab : 0.f) + accf[A_CV + c];
  ph[i] = v;
}

// ---------------- background BCE+Dice partial sums ----------------
__global__ void cl_bg(const float* __restrict__ pred, const int* __restrict__ tm,
                      const unsigned* __restrict__ st, unsigned* acc) {
  int i = blockIdx.x * 256 + threadIdx.x;
  int b = i >> 18; int r = i & (HW - 1); int y = r >> 9; int x = r & 511;
  unsigned s = st[((size_t)b * WP + y + 1) * WP + x + 1];
  float p = (s >> 24) ? 0.0f : pred[(size_t)b * NC * HW + r];  // channel 0
  float t = (tm[i] == 0) ? 1.f : 0.f;
  const float EPS = 1e-7f;
  float pc = fminf(fmaxf(p, EPS), 1.0f - EPS);
  float bce = t * logf(pc) + (1.f - t) * logf(1.f - pc);
  float s0 = wave_red(bce), s1 = wave_red(p * t), s2 = wave_red(p), s3 = wave_red(t);
  if ((threadIdx.x & 31) == 0) {
    float* accf = (float*)acc;
    atomicAdd(&accf[A_BG + 0], s0); atomicAdd(&accf[A_BG + 1], s1);
    atomicAdd(&accf[A_BG + 2], s2); atomicAdd(&accf[A_BG + 3], s3);
  }
}

// ---------------- median radix-select: 3 histogram passes ----------------
__global__ void cl_histA(const float* __restrict__ ph, const int* __restrict__ tm,
                         unsigned* acc) {
  int i = blockIdx.x * 256 + threadIdx.x;
  int t = tm[i];
  atomicAdd(&acc[A_CNT_T + t], 1u);
  if (t > 0) {
    unsigned u = __float_as_uint(ph[i]);   // ph >= 0: bit order == value order
    atomicAdd(&acc[A_HA + (t - 1) * 2048 + (u >> 21)], 1u);
  }
}

__global__ void cl_histB(const float* __restrict__ ph, const int* __restrict__ tm,
                         unsigned* acc) {
  int i = blockIdx.x * 256 + threadIdx.x;
  int t = tm[i];
  if (t > 0) {
    unsigned u = __float_as_uint(ph[i]);
    if ((u >> 21) == acc[A_B1 + t])
      atomicAdd(&acc[A_HB + (t - 1) * 2048 + ((u >> 10) & 2047u)], 1u);
  }
}

__global__ void cl_histC(const float* __restrict__ ph, const int* __restrict__ tm,
                         unsigned* acc) {
  int i = blockIdx.x * 256 + threadIdx.x;
  int t = tm[i];
  if (t > 0) {
    unsigned u = __float_as_uint(ph[i]);
    if ((u >> 10) == ((acc[A_B1 + t] << 11) | acc[A_B2 + t]))
      atomicAdd(&acc[A_HC + (t - 1) * 1024 + (u & 1023u)], 1u);
  }
}

__global__ void cl_select(unsigned* acc, int stage) {
  int t = (int)threadIdx.x + 1;
  if (t > 7 || blockIdx.x) return;
  unsigned n = acc[A_CNT_T + t];
  if (!n) {
    if (stage == 0) acc[A_MED + t] = 0x7f800000u;  // +inf: matches all-inf median
    return;
  }
  if (stage == 0) {
    unsigned k = (n - 1u) >> 1;                    // torch lower median rank
    const unsigned* h = &acc[A_HA + (t - 1) * 2048];
    unsigned csum = 0;
    for (int bin = 0; bin < 2048; ++bin) {
      unsigned c = h[bin];
      if (csum + c > k) { acc[A_B1 + t] = (unsigned)bin; acc[A_RK1 + t] = k - csum; return; }
      csum += c;
    }
  } else if (stage == 1) {
    unsigned k = acc[A_RK1 + t];
    const unsigned* h = &acc[A_HB + (t - 1) * 2048];
    unsigned csum = 0;
    for (int bin = 0; bin < 2048; ++bin) {
      unsigned c = h[bin];
      if (csum + c > k) { acc[A_B2 + t] = (unsigned)bin; acc[A_RK2 + t] = k - csum; return; }
      csum += c;
    }
  } else {
    unsigned k = acc[A_RK2 + t];
    const unsigned* h = &acc[A_HC + (t - 1) * 1024];
    unsigned csum = 0;
    for (int bin = 0; bin < 1024; ++bin) {
      unsigned c = h[bin];
      if (csum + c > k) {
        acc[A_MED + t] = (acc[A_B1 + t] << 21) | (acc[A_B2 + t] << 10) | (unsigned)bin;
        return;
      }
      csum += c;
    }
  }
}

// ---------------- per-target confusion counts via WMMA ----------------
// For each 32-pixel chunk: A(16x32 f16) rows = [ones, f_1..f_7, 0...],
// B(32x16 f16) cols = [ones, tv_1..tv_7, 0...].  D = A*B + C accumulates
// D[t][t]=Sum f_t*tv_t, D[t][0]=Sum f_t, D[0][t]=Sum tv_t — exact (0/1 in f16,
// f32 accumulate).  BCE is derived from these counts in cl_final.
__global__ void cl_med(const float* __restrict__ ph, const int* __restrict__ tm,
                       unsigned* acc) {
  const int lane = threadIdx.x & 31;
  const int wave = (blockIdx.x * 256 + threadIdx.x) >> 5;   // 4096 waves
  float* accf = (float*)acc;

  float med[8];
  #pragma unroll
  for (int t = 1; t < 8; ++t) med[t] = __uint_as_float(acc[A_MED + t]);

  v8f d = {0.f, 0.f, 0.f, 0.f, 0.f, 0.f, 0.f, 0.f};
  const int half = (lane >= 16) ? 8 : 0;
  const int row = lane & 15;

  for (int j = 0; j < 8; ++j) {                  // 8 chunks of 32 pixels per wave
    int pix = ((wave * 8 + j) << 5) + lane;
    float p = ph[pix];
    int tv = tm[pix];
    unsigned am = 1u;                            // bit0 = ones row
    #pragma unroll
    for (int t = 1; t < 8; ++t) am |= (p == med[t]) ? (1u << t) : 0u;
    unsigned bm = (1u << tv) | 1u;               // bit0 = ones col, bit tv = indicator

    v16h A, B;
    #pragma unroll
    for (int e = 0; e < 16; ++e) {
      int K = ((e < 8) ? e : e + 8) + half;      // ISA 16-bit A 16x32 K mapping
      unsigned amk = __shfl(am, K, 32);
      unsigned bmk = __shfl(bm, K, 32);
      A[e] = (_Float16)(float)((amk >> row) & 1u);
      B[e] = (_Float16)(float)((bmk >> row) & 1u);
    }
    d = __builtin_amdgcn_wmma_f32_16x16x32_f16(false, A, false, B,
                                               (short)0, d, false, false);
  }

  // scatter: D layout — lane n (0..15) holds column n; VGPR r holds row r (lanes<16)
  if (lane >= 1 && lane <= 7) {
    atomicAdd(&accf[A_TSUM + (lane - 1) * 4 + 1], d[lane]);  // S_ftv[t] = D[t][t]
    atomicAdd(&accf[A_TSUM + (lane - 1) * 4 + 3], d[0]);     // S_tv[t]  = D[0][t]
  }
  if (lane == 0) {
    for (int t = 1; t < 8; ++t)
      atomicAdd(&accf[A_TSUM + (t - 1) * 4 + 2], d[t]);      // S_f[t]   = D[t][0]
  }
}

// ---------------- final combine ----------------
__global__ void cl_final(const unsigned* __restrict__ acc, float* __restrict__ out) {
  if (threadIdx.x || blockIdx.x) return;
  const float* accf = (const float*)acc;
  const float N = (float)TOT;
  const float EPS = 1e-7f;
  const float L0 = logf(EPS);
  const float L1 = logf(1.0f - EPS);
  float res = -(accf[A_BG + 0] / N)
            + 1.f - (2.f * accf[A_BG + 1] + 1.f) / (accf[A_BG + 2] + accf[A_BG + 3] + 1.f);
  for (int t = 1; t < 8; ++t) {
    if (acc[A_CNT_T + t]) {
      const float* s = &accf[A_TSUM + (t - 1) * 4];
      float sftv = s[1], sf = s[2], stv = s[3];
      float ndis = sf + stv - 2.f * sftv;        // pixels where f != tv
      float bsum = L1 * (N - ndis) + L0 * ndis;  // sum of per-pixel BCE terms
      res += -(bsum / N) + 1.f - (2.f * sftv + 1.f) / (sf + stv + 1.f);
    }
  }
  int nu = 0;
  for (int t = 0; t < 8; ++t) nu += (acc[A_CNT_T + t] > 0u) ? 1 : 0;
  out[0] = res / ((float)nu + 1.f);
}

// ---------------- launcher ----------------
extern "C" void kernel_launch(void* const* d_in, const int* in_sizes, int n_in,
                              void* d_out, int out_size, void* d_ws, size_t ws_size,
                              hipStream_t stream) {
  const float* pred  = (const float*)d_in[0];   // [4,8,512,512] f32
  const int*   tmask = (const int*)d_in[1];     // [4,1,512,512] i32
  float* out = (float*)d_out;

  uint8_t* ws = (uint8_t*)d_ws;
  unsigned* st  = (unsigned*)ws;                                   // padded state
  float*    ph  = (float*)(ws + (size_t)BATCH * PLANEP * 4);       // placeholder
  unsigned* acc = (unsigned*)(ws + (size_t)BATCH * PLANEP * 4 + (size_t)TOT * 4);

  cl_zero<<<(A_END + 255) / 256, 256, 0, stream>>>(acc);
  cl_init<<<(BATCH * PLANEP + 255) / 256, 256, 0, stream>>>(pred, st);
  for (int p = 0; p < CC_PASSES; ++p)
    cl_cc<<<dim3(16, 16, BATCH), 256, 0, stream>>>(st);
  cl_stats<<<TOT / 256, 256, 0, stream>>>(st, acc);
  cl_scalars<<<1, 32, 0, stream>>>(acc);
  cl_place<<<TOT / 256, 256, 0, stream>>>(st, acc, ph);
  cl_bg<<<TOT / 256, 256, 0, stream>>>(pred, tmask, st, acc);
  cl_histA<<<TOT / 256, 256, 0, stream>>>(ph, tmask, acc);
  cl_select<<<1, 32, 0, stream>>>(acc, 0);
  cl_histB<<<TOT / 256, 256, 0, stream>>>(ph, tmask, acc);
  cl_select<<<1, 32, 0, stream>>>(acc, 1);
  cl_histC<<<TOT / 256, 256, 0, stream>>>(ph, tmask, acc);
  cl_select<<<1, 32, 0, stream>>>(acc, 2);
  cl_med<<<MED_BLOCKS, 256, 0, stream>>>(ph, tmask, acc);
  cl_final<<<1, 32, 0, stream>>>(acc, out);
}
